// GrapeModule_45518063403048
// MI455X (gfx1250) — matrix operand order repaired and checked by reading, not compiled, not run on hardware
//
#include <hip/hip_runtime.h>
#include <hip/hip_bf16.h>
#include <stdint.h>

// ---------------------------------------------------------------------------
// GRAPE GNN forward on MI455X (gfx1250), bf16 WMMA with f32 accumulation.
// ---------------------------------------------------------------------------

typedef __attribute__((ext_vector_type(16))) __bf16 v16bf;
typedef __attribute__((ext_vector_type(8)))  float  v8f;
typedef __attribute__((ext_vector_type(4)))  float  v4f;

#define NNODES 50000
#define NEDGES 800000

// Generic fused linear:
//   out[row, :] = act( sum_seg  SRCseg[gather(row)] @ W[kb_seg:kb_seg+len, :]  + bias )
// Each wave computes a 16-row x OUTD tile with v_wmma_f32_16x16x32_bf16.
// Weights are transposed+converted to bf16 in LDS once per workgroup.
// SCATTER=true: atomically accumulate rows into out[oidx[row], :] (segment_sum).
template <int KTOT, int OUTD, bool RELU, bool SCATTER>
__global__ __launch_bounds__(256) void wmma_linear(
    const float* __restrict__ W,     // [KTOT, OUTD] row-major f32
    const float* __restrict__ bias,  // [OUTD]
    const float* __restrict__ b0, const long long* __restrict__ g0, int ld0, int kb0,
    const float* __restrict__ b1, const long long* __restrict__ g1, int ld1, int kb1,
    const float* __restrict__ b2, const long long* __restrict__ g2, int ld2, int kb2,
    float* __restrict__ out, const long long* __restrict__ oidx, int oldim,
    int nrows)
{
    constexpr int NT = OUTD / 16;
    // Transposed bf16 weights: Wt[n][k], contiguous in k for vector B loads.
    __shared__ __bf16 Wt[OUTD * KTOT];
    for (int i = threadIdx.x; i < KTOT * OUTD; i += 256) {
        int k = i / OUTD, n = i % OUTD;
        Wt[n * KTOT + k] = (__bf16)W[i];
    }
    __syncthreads();

    const int wave = threadIdx.x >> 5;
    const int lane = threadIdx.x & 31;
    const int tile = blockIdx.x * 8 + wave;
    if (tile * 16 >= nrows) return;

    const int m    = lane & 15;   // row within tile (A) / column (B, C/D)
    const int half = lane >> 4;   // lane half selects K sub-range / row offset

    const int row  = tile * 16 + m;
    const int rowc = row < nrows ? row : nrows - 1;

    // Pre-resolve gather indices per segment.
    const long long g0v = b0 ? (g0 ? g0[rowc] : (long long)rowc) : 0;
    const long long g1v = b1 ? (g1 ? g1[rowc] : (long long)rowc) : 0;
    const long long g2v = b2 ? (g2 ? g2[rowc] : (long long)rowc) : 0;

    v8f acc[NT];
#pragma unroll
    for (int nt = 0; nt < NT; ++nt)
#pragma unroll
        for (int r = 0; r < 8; ++r) acc[nt][r] = 0.0f;

#pragma unroll
    for (int k0 = 0; k0 < KTOT; k0 += 32) {
        // Select K-segment (segment boundaries are multiples of 32).
        const float* base;
        int kb;
        if (b2 && k0 >= kb2)      { base = b2 + g2v * (long long)ld2; kb = kb2; }
        else if (b1 && k0 >= kb1) { base = b1 + g1v * (long long)ld1; kb = kb1; }
        else                      { base = b0 + g0v * (long long)ld0; kb = kb0; }

        // A fragment (16x32 bf16): lane half h holds K = k0+16h+0..15.
        const float* p = base + (k0 - kb) + half * 16;
        v4f f0 = *(const v4f*)(p);
        v4f f1 = *(const v4f*)(p + 4);
        v4f f2 = *(const v4f*)(p + 8);
        v4f f3 = *(const v4f*)(p + 12);
        v16bf a;
#pragma unroll
        for (int i = 0; i < 4; ++i) {
            a[i]      = (__bf16)f0[i];
            a[4 + i]  = (__bf16)f1[i];
            a[8 + i]  = (__bf16)f2[i];
            a[12 + i] = (__bf16)f3[i];
        }

#pragma unroll
        for (int nt = 0; nt < NT; ++nt) {
            // B fragment (32x16 bf16): lane = column n; low lanes K=k0..k0+15,
            // high lanes K=k0+16..k0+31 -> one contiguous 32-byte LDS read.
            const v16bf b =
                *(const v16bf*)&Wt[(nt * 16 + m) * KTOT + k0 + half * 16];
            acc[nt] = __builtin_amdgcn_wmma_f32_16x16x32_bf16(
                false, a, false, b, (short)0, acc[nt], false, false);
        }
    }

    // Epilogue: bias + activation + (scatter-)store.
    // C/D layout: VGPR r -> row (r + 8*half), lane -> column (m).
#pragma unroll
    for (int nt = 0; nt < NT; ++nt) {
        const int n = nt * 16 + m;
        const float bv = bias[n];
#pragma unroll
        for (int r = 0; r < 8; ++r) {
            float v = acc[nt][r] + bv;
            if (RELU) v = fmaxf(v, 0.0f);
            const int orow = tile * 16 + r + half * 8;
            if (orow < nrows) {
                if (SCATTER) {
                    const long long go = oidx[orow];
                    atomicAdd(out + go * (long long)oldim + n, v);
                } else {
                    out[(long long)orow * oldim + n] = v;
                }
            }
        }
    }
}

// Final edge predictor: out[e] = h[src]@W[0:128] + h[dst]@W[128:256]
//                               + ea[e]@W[256:320] + b.  (OUT dim = 1)
__global__ __launch_bounds__(256) void epred_kernel(
    const float* __restrict__ h, const float* __restrict__ ea,
    const long long* __restrict__ src, const long long* __restrict__ dst,
    const float* __restrict__ W, const float* __restrict__ bias,
    float* __restrict__ out, int E)
{
    __shared__ float w[320];
    for (int i = threadIdx.x; i < 320; i += 256) w[i] = W[i];
    __syncthreads();

    const int e = blockIdx.x * 256 + threadIdx.x;
    if (e >= E) return;
    const long long s = src[e], d = dst[e];
    const float* hs = h + s * 128;
    const float* hd = h + d * 128;
    const float* ep = ea + (long long)e * 64;
    float acc = bias[0];
#pragma unroll 4
    for (int k = 0; k < 128; ++k) acc += hs[k] * w[k] + hd[k] * w[128 + k];
#pragma unroll 4
    for (int k = 0; k < 64; ++k) acc += ep[k] * w[256 + k];
    out[e] = acc;
}

extern "C" void kernel_launch(void* const* d_in, const int* in_sizes, int n_in,
                              void* d_out, int out_size, void* d_ws, size_t ws_size,
                              hipStream_t stream)
{
    (void)in_sizes; (void)n_in; (void)out_size; (void)ws_size;

    const float*     x       = (const float*)d_in[0];
    const float*     ea_in   = (const float*)d_in[1];
    const long long* ei      = (const long long*)d_in[2];   // int64 [2, E]
    const float*     emb_W   = (const float*)d_in[3];
    const float*     emb_b   = (const float*)d_in[4];
    const float*     msg_W   = (const float*)d_in[5];       // [2,192,128]
    const float*     msg_b   = (const float*)d_in[6];       // [2,128]
    const float*     upd_W   = (const float*)d_in[7];       // [2,256,128]
    const float*     upd_b   = (const float*)d_in[8];       // [2,128]
    const float*     eupd_W  = (const float*)d_in[9];       // [2,320,64]
    const float*     eupd_b  = (const float*)d_in[10];      // [2,64]
    const float*     npred_W = (const float*)d_in[11];      // [128,32]
    const float*     npred_b = (const float*)d_in[12];
    const float*     epred_W = (const float*)d_in[13];      // [320,1]
    const float*     epred_b = (const float*)d_in[14];

    const long long* src = ei;            // edge_index[0]
    const long long* dst = ei + NEDGES;   // edge_index[1]

    // Workspace layout (f32): h[N,128] | aggr[N,128] | ea[E,64]  (~257 MB)
    float* h    = (float*)d_ws;
    float* aggr = h + (size_t)NNODES * 128;
    float* ea   = aggr + (size_t)NNODES * 128;

    float* node_pred = (float*)d_out;                       // [N,32]
    float* edge_pred = node_pred + (size_t)NNODES * 32;     // [E,1]

    const dim3 blk(256);
    const int gridN = (NNODES + 127) / 128;  // 8 waves/block, 16 rows/wave
    const int gridE = (NEDGES + 127) / 128;
    const int KBIG  = 1 << 30;

    // h = x @ emb_W + emb_b
    wmma_linear<64, 128, false, false><<<gridN, blk, 0, stream>>>(
        emb_W, emb_b,
        x, nullptr, 64, 0,
        nullptr, nullptr, 0, KBIG,
        nullptr, nullptr, 0, KBIG,
        h, nullptr, 128, NNODES);

    const float* ea_cur = ea_in;
    for (int l = 0; l < 2; ++l) {
        hipMemsetAsync(aggr, 0, (size_t)NNODES * 128 * sizeof(float), stream);

        // aggr[dst] += relu( h[src]@msgW[:128] + ea@msgW[128:] + msg_b )
        wmma_linear<192, 128, true, true><<<gridE, blk, 0, stream>>>(
            msg_W + (size_t)l * 192 * 128, msg_b + (size_t)l * 128,
            h,      src,     128, 0,
            ea_cur, nullptr, 64,  128,
            nullptr, nullptr, 0,  KBIG,
            aggr, dst, 128, NEDGES);

        // h = relu( aggr@updW[:128] + h@updW[128:] + upd_b )   (in-place safe)
        wmma_linear<256, 128, true, false><<<gridN, blk, 0, stream>>>(
            upd_W + (size_t)l * 256 * 128, upd_b + (size_t)l * 128,
            aggr, nullptr, 128, 0,
            h,    nullptr, 128, 128,
            nullptr, nullptr, 0, KBIG,
            h, nullptr, 128, NNODES);

        // ea = relu( h[src]@W[:128] + h[dst]@W[128:256] + ea@W[256:] + b )
        wmma_linear<320, 64, true, false><<<gridE, blk, 0, stream>>>(
            eupd_W + (size_t)l * 320 * 64, eupd_b + (size_t)l * 64,
            h,      src,     128, 0,
            h,      dst,     128, 128,
            ea_cur, nullptr, 64,  256,
            ea, nullptr, 64, NEDGES);
        ea_cur = ea;  // layer 1 updates in place (row-wise independent)
    }

    // node_pred = h @ npred_W + npred_b
    wmma_linear<128, 32, false, false><<<gridN, blk, 0, stream>>>(
        npred_W, npred_b,
        h, nullptr, 128, 0,
        nullptr, nullptr, 0, KBIG,
        nullptr, nullptr, 0, KBIG,
        node_pred, nullptr, 32, NNODES);

    // edge_pred (scalar output per edge): plain VALU dot products.
    epred_kernel<<<(NEDGES + 255) / 256, blk, 0, stream>>>(
        h, ea, src, dst, epred_W, epred_b, edge_pred, NEDGES);
}
